// LatticeTokenPtrNet_48490180771920
// MI455X (gfx1250) — compile-verified, matrix operand order checked by reading, not run on hardware
//
#include <hip/hip_runtime.h>
#include <hip/hip_bf16.h>
#include <math.h>

// ---------------------------------------------------------------------------
// LatticeTokenPtrNet for MI455X (gfx1250, wave32, WMMA)
//  - All input-side projections hoisted into f16 WMMA GEMMs (f32 accumulate):
//      Gx0 = X @ Wih0^T, Gx1 = Y @ Wih1^T  (per direction),
//      K   = enc @ Wk,   Gd0 = DecIn @ decWih0^T  (teacher forcing makes the
//      whole decoder input sequence known up front).
//  - Sequential LSTM recurrences run as persistent 1024-thread workgroups with
//    LDS-resident h/c state. Recurrent weights are pre-transposed so every
//    inner-product step is a fully coalesced 4KB line from L2 + LDS broadcast.
// ---------------------------------------------------------------------------

typedef __attribute__((ext_vector_type(16))) _Float16 v16h;
typedef __attribute__((ext_vector_type(8)))  float    v8f;

#define T_STEPS   64
#define A_NUM     16
#define SEQ       (T_STEPS * A_NUM)   // 1024
#define E_DIM     256
#define H_ENC     256
#define G_ENC     (4 * H_ENC)         // 1024
#define DEC_HID   512
#define G_DEC     (4 * DEC_HID)       // 2048
#define D_ATTN    256

__device__ __forceinline__ float sigf(float x) { return 1.0f / (1.0f + __expf(-x)); }

// ------------------------- elementwise converts ----------------------------
__global__ void k_f32_to_f16(const float* __restrict__ s, _Float16* __restrict__ d, int n) {
  int i = blockIdx.x * blockDim.x + threadIdx.x;
  if (i < n) d[i] = (_Float16)s[i];
}

// src [R][C] f32 -> dst [C][R] f16  (attn_Wk so the WMMA GEMM sees W=[N,K])
__global__ void k_transpose_f16(const float* __restrict__ src, _Float16* __restrict__ dst,
                                int R, int C) {
  int i = blockIdx.x * blockDim.x + threadIdx.x;
  if (i >= R * C) return;
  int r = i / C, c = i % C;
  dst[(size_t)c * R + r] = (_Float16)src[i];
}

// src [R][C] f32 -> dst [C][R] f32  (recurrent weights: coalesce the scans)
__global__ void k_transpose_f32(const float* __restrict__ src, float* __restrict__ dst,
                                int R, int C) {
  int i = blockIdx.x * blockDim.x + threadIdx.x;
  if (i >= R * C) return;
  int r = i / C, c = i % C;
  dst[(size_t)c * R + r] = src[i];
}

// ------------------------- embedding gathers -------------------------------
// lattice [T,A,4] int -> X f16 [1024, 256]  (4 features x 64-dim embeds)
__global__ void k_embed_lattice(const int* __restrict__ lattice,
                                const float* __restrict__ tab,
                                _Float16* __restrict__ Xh) {
  int i = blockIdx.x * blockDim.x + threadIdx.x;
  if (i >= SEQ * E_DIM) return;
  int r = i >> 8;           // 0..1023 = t*16 + a
  int cidx = i & 255;
  int f = cidx >> 6, d = cidx & 63;
  int tok = lattice[r * 4 + f];
  Xh[i] = (_Float16)tab[(size_t)tok * 64 + d];
}

// Decoder input seq (teacher forced, known up front): [64][512] f16
// row t = concat( prev_emb(t), tok_emb(t) )
__global__ void k_build_dec_in(const int* __restrict__ inputs, const int* __restrict__ gold,
                               const int* __restrict__ sos,
                               const float* __restrict__ lat_tab,
                               const float* __restrict__ in_tab,
                               const _Float16* __restrict__ Xh,
                               _Float16* __restrict__ Inp0) {
  int i = blockIdx.x * blockDim.x + threadIdx.x;
  if (i >= T_STEPS * 512) return;
  int t = i >> 9;
  int c = i & 511;
  float v;
  if (c < 256) {
    if (t == 0) {
      int f = c >> 6, d = c & 63;
      v = lat_tab[(size_t)sos[f] * 64 + d];
    } else {
      v = (float)Xh[(size_t)((t - 1) * A_NUM + gold[t - 1]) * E_DIM + c];
    }
  } else {
    v = in_tab[(size_t)inputs[t] * 256 + (c - 256)];
  }
  Inp0[i] = (_Float16)v;
}

// ------------------------- WMMA GEMM ---------------------------------------
// C[M,N] = A[M,K] @ W[N,K]^T (+bias[N]).  f16 in, f32 out.
// One wave per 16x16 output tile; K in chunks of 32 via v_wmma_f32_16x16x32_f16.
// Fragment K-index map per ISA 7.12.2 (16-bit A 16x32, wave32):
//   lane L: row = L%16, hs = L/16; elem e (vgpr j=e/2):
//   k = 2j + (j>=4 ? 8 : 0) + 8*hs + (e&1)
// (two contiguous 8-half runs -> compiler emits paired global_load_b128)
__device__ __forceinline__ v16h load_frag(const _Float16* __restrict__ row, int k0, int hs) {
  v16h f;
#pragma unroll
  for (int e = 0; e < 16; ++e) {
    int j = e >> 1;
    int k = k0 + 2 * j + ((j >= 4) ? 8 : 0) + 8 * hs + (e & 1);
    f[e] = row[k];
  }
  return f;
}

__global__ void k_wmma_gemm(const _Float16* __restrict__ A, const _Float16* __restrict__ W,
                            const float* __restrict__ bias, float* __restrict__ C,
                            int M, int N, int K) {
  int gtid = blockIdx.x * blockDim.x + threadIdx.x;
  int wave = gtid >> 5;
  int lane = gtid & 31;
  int tilesN = N >> 4;
  int totalTiles = (M >> 4) * tilesN;
  if (wave >= totalTiles) return;          // wave-uniform: EXEC all-ones inside
  int tm = (wave / tilesN) << 4;
  int tn = (wave % tilesN) << 4;
  int r  = lane & 15;
  int hs = lane >> 4;
  const _Float16* Arow = A + (size_t)(tm + r) * K;
  const _Float16* Wrow = W + (size_t)(tn + r) * K;   // B[k][n] = W[n][k]
  v8f acc = {};
  for (int k0 = 0; k0 < K; k0 += 32) {
    v16h af = load_frag(Arow, k0, hs);
    v16h bf = load_frag(Wrow, k0, hs);
    acc = __builtin_amdgcn_wmma_f32_16x16x32_f16(false, af, false, bf,
                                                 (short)0, acc, false, false);
  }
  int n = tn + r;
  float bv = bias ? bias[n] : 0.0f;
#pragma unroll
  for (int rr = 0; rr < 8; ++rr) {
    C[(size_t)(tm + rr + 8 * hs) * N + n] = acc[rr] + bv;  // M = rr + 8*hs, N = lane%16
  }
}

// ------------------------- encoder recurrent scan --------------------------
// blockIdx.x = direction (0 fwd, 1 bwd).  1024 threads = 4H gate outputs.
// Gx:   [2][SEQ][1024] precomputed x@Wih^T + b.
// WhhT: [2][256][1024]  (transposed: column n coalesced across the block)
// yout: [SEQ][512] f16 (fwd cols 0..255, bwd 256..511).
__global__ void k_enc_scan(const float* __restrict__ Gx, const float* __restrict__ WhhT,
                           _Float16* __restrict__ yout,
                           float* __restrict__ hfin, float* __restrict__ cfin) {
  const int dir = blockIdx.x;
  const int n = threadIdx.x;
  __shared__ float h[H_ENC], c[H_ENC], g[G_ENC];
  const float* GxD = Gx + (size_t)dir * SEQ * G_ENC;
  const float* Wt  = WhhT + (size_t)dir * H_ENC * G_ENC;
  if (n < H_ENC) { h[n] = 0.0f; c[n] = 0.0f; }
  __syncthreads();
  for (int s = 0; s < SEQ; ++s) {
    int t = dir ? (SEQ - 1 - s) : s;
    float acc = GxD[(size_t)t * G_ENC + n];
#pragma unroll 8
    for (int j = 0; j < H_ENC; ++j) acc += h[j] * Wt[(size_t)j * G_ENC + n];
    g[n] = acc;
    __syncthreads();
    if (n < H_ENC) {
      float ig = sigf(g[n]);
      float fg = sigf(g[H_ENC + n]);
      float gg = tanhf(g[2 * H_ENC + n]);
      float og = sigf(g[3 * H_ENC + n]);
      float cn = fg * c[n] + ig * gg;
      c[n] = cn;
      float hn = og * tanhf(cn);
      h[n] = hn;
      yout[(size_t)t * 512 + dir * H_ENC + n] = (_Float16)hn;
    }
    __syncthreads();
  }
  if (n < H_ENC) { hfin[dir * H_ENC + n] = h[n]; cfin[dir * H_ENC + n] = c[n]; }
}

// ------------------------- decoder scan + pointer attention ----------------
// All recurrent / layer-1 input weights pre-transposed to [512][2048]:
// per j-step the block reads one coalesced 8KB line, h[j] is an LDS broadcast.
__global__ void k_decoder(const float* __restrict__ Gd0,    // [64][2048] x@Wih0^T + b0
                          const float* __restrict__ Wi1T,   // [512][2048] layer1 input^T
                          const float* __restrict__ WhhT,   // [2][512][2048] recurrent^T
                          const float* __restrict__ b1,     // [2048] layer1 bias
                          const float* __restrict__ Kmat,   // [1024][256] enc@Wk
                          const float* __restrict__ Wq,     // [512][256]
                          const float* __restrict__ av,     // [256]
                          const float* __restrict__ hinit,  // [2][512]
                          const float* __restrict__ cinit,
                          const int* __restrict__ alens,    // [64]
                          float* __restrict__ out) {        // [64][16]
  const int tid = threadIdx.x;
  __shared__ float h0[DEC_HID], c0[DEC_HID], h1[DEC_HID], c1[DEC_HID];
  __shared__ float g[G_DEC], q[D_ATTN];
  if (tid < DEC_HID) {
    h0[tid] = hinit[tid];           c0[tid] = cinit[tid];
    h1[tid] = hinit[DEC_HID + tid]; c1[tid] = cinit[DEC_HID + tid];
  }
  __syncthreads();
  const float* Wt0 = WhhT;                              // layer0 recurrent^T
  const float* Wt1 = WhhT + (size_t)DEC_HID * G_DEC;    // layer1 recurrent^T
  const int n0 = tid, n1 = tid + 1024;
  for (int t = 0; t < T_STEPS; ++t) {
    // layer 0: g = Gd0[t] + h0 @ Whh0^T   (2 gate outputs per thread)
    {
      float a0 = Gd0[(size_t)t * G_DEC + n0];
      float a1 = Gd0[(size_t)t * G_DEC + n1];
#pragma unroll 4
      for (int j = 0; j < DEC_HID; ++j) {
        float hj = h0[j];
        const float* row = Wt0 + (size_t)j * G_DEC;
        a0 += hj * row[n0];
        a1 += hj * row[n1];
      }
      g[n0] = a0; g[n1] = a1;
    }
    __syncthreads();
    if (tid < DEC_HID) {
      float ig = sigf(g[tid]);
      float fg = sigf(g[DEC_HID + tid]);
      float gg = tanhf(g[2 * DEC_HID + tid]);
      float og = sigf(g[3 * DEC_HID + tid]);
      float cn = fg * c0[tid] + ig * gg;
      c0[tid] = cn;
      h0[tid] = og * tanhf(cn);
    }
    __syncthreads();
    // layer 1: g = b1 + h0 @ Wih1^T + h1 @ Whh1^T
    {
      float a0 = b1[n0];
      float a1 = b1[n1];
#pragma unroll 2
      for (int j = 0; j < DEC_HID; ++j) {
        float xj = h0[j], yj = h1[j];
        const float* ri = Wi1T + (size_t)j * G_DEC;
        const float* rr = Wt1 + (size_t)j * G_DEC;
        a0 += xj * ri[n0] + yj * rr[n0];
        a1 += xj * ri[n1] + yj * rr[n1];
      }
      g[n0] = a0; g[n1] = a1;
    }
    __syncthreads();
    if (tid < DEC_HID) {
      float ig = sigf(g[tid]);
      float fg = sigf(g[DEC_HID + tid]);
      float gg = tanhf(g[2 * DEC_HID + tid]);
      float og = sigf(g[3 * DEC_HID + tid]);
      float cn = fg * c1[tid] + ig * gg;
      c1[tid] = cn;
      h1[tid] = og * tanhf(cn);
    }
    __syncthreads();
    // attention query q = h1 @ Wq  (Wq[j][d]: coalesced across tid)
    if (tid < D_ATTN) {
      float acc = 0.0f;
#pragma unroll 8
      for (int j = 0; j < DEC_HID; ++j) acc += h1[j] * Wq[(size_t)j * D_ATTN + tid];
      q[tid] = acc;
    }
    __syncthreads();
    // scores[a] = sum_d tanh(K[t,a,d] + q[d]) * v[d], masked by analysis length
    if (tid < A_NUM) {
      const float* Kr = Kmat + (size_t)(t * A_NUM + tid) * D_ATTN;
      float s = 0.0f;
      for (int d = 0; d < D_ATTN; ++d) s += tanhf(Kr[d] + q[d]) * av[d];
      out[t * A_NUM + tid] = (tid < alens[t]) ? s : -1e10f;
    }
    __syncthreads();
  }
}

// ---------------------------------------------------------------------------
extern "C" void kernel_launch(void* const* d_in, const int* in_sizes, int n_in,
                              void* d_out, int out_size, void* d_ws, size_t ws_size,
                              hipStream_t stream) {
  const int*   lattice = (const int*)d_in[0];
  const int*   alens   = (const int*)d_in[1];
  const int*   inputs  = (const int*)d_in[2];
  const int*   gold    = (const int*)d_in[4];
  const int*   sos     = (const int*)d_in[5];
  const float* lat_tab = (const float*)d_in[6];
  const float* in_tab  = (const float*)d_in[7];
  const float* eWih0   = (const float*)d_in[8];
  const float* eWhh0   = (const float*)d_in[9];
  const float* eb0     = (const float*)d_in[10];
  const float* eWih1   = (const float*)d_in[11];
  const float* eWhh1   = (const float*)d_in[12];
  const float* eb1     = (const float*)d_in[13];
  const float* dWih    = (const float*)d_in[14];
  const float* dWhh    = (const float*)d_in[15];
  const float* db      = (const float*)d_in[16];
  const float* Wq      = (const float*)d_in[17];
  const float* Wk      = (const float*)d_in[18];
  const float* av      = (const float*)d_in[19];
  float* out = (float*)d_out;

  char* ws = (char*)d_ws;
  size_t off = 0;
  auto wsalloc = [&](size_t bytes) -> void* {
    void* p = (void*)(ws + off);
    off += (bytes + 255) & ~(size_t)255;
    return p;
  };

  _Float16* Xh     = (_Float16*)wsalloc((size_t)SEQ * E_DIM * 2);        // lattice embeds f16
  _Float16* Yh     = (_Float16*)wsalloc((size_t)SEQ * 512 * 2);          // enc layer0 out
  _Float16* Ench   = (_Float16*)wsalloc((size_t)SEQ * 512 * 2);          // enc layer1 out
  _Float16* Inp0   = (_Float16*)wsalloc((size_t)T_STEPS * 512 * 2);      // decoder inputs
  _Float16* Wih0h  = (_Float16*)wsalloc((size_t)2 * G_ENC * E_DIM * 2);
  _Float16* Wih1h  = (_Float16*)wsalloc((size_t)2 * G_ENC * 512 * 2);
  _Float16* dWih0h = (_Float16*)wsalloc((size_t)G_DEC * DEC_HID * 2);
  _Float16* WkTh   = (_Float16*)wsalloc((size_t)D_ATTN * DEC_HID * 2);
  float*    WhhT0  = (float*)wsalloc((size_t)2 * H_ENC * G_ENC * 4);     // enc rec^T
  float*    WhhT1  = (float*)wsalloc((size_t)2 * H_ENC * G_ENC * 4);
  float*    dWi1T  = (float*)wsalloc((size_t)DEC_HID * G_DEC * 4);       // dec layer1 in^T
  float*    dWhhT  = (float*)wsalloc((size_t)2 * DEC_HID * G_DEC * 4);   // dec rec^T
  float*    Gx0    = (float*)wsalloc((size_t)2 * SEQ * G_ENC * 4);
  float*    Gx1    = (float*)wsalloc((size_t)2 * SEQ * G_ENC * 4);
  float*    Gd0    = (float*)wsalloc((size_t)T_STEPS * G_DEC * 4);
  float*    Kmat   = (float*)wsalloc((size_t)SEQ * D_ATTN * 4);
  float*    hinit  = (float*)wsalloc((size_t)2 * DEC_HID * 4);
  float*    cinit  = (float*)wsalloc((size_t)2 * DEC_HID * 4);
  (void)ws_size; (void)in_sizes; (void)n_in; (void)out_size;

  auto blocks = [](int n) { return (n + 255) / 256; };

  // 1. embeds + weight conversion / transposition
  k_embed_lattice<<<blocks(SEQ * E_DIM), 256, 0, stream>>>(lattice, lat_tab, Xh);
  k_f32_to_f16<<<blocks(2 * G_ENC * E_DIM), 256, 0, stream>>>(eWih0, Wih0h, 2 * G_ENC * E_DIM);
  k_f32_to_f16<<<blocks(2 * G_ENC * 512), 256, 0, stream>>>(eWih1, Wih1h, 2 * G_ENC * 512);
  k_f32_to_f16<<<blocks(G_DEC * DEC_HID), 256, 0, stream>>>(dWih, dWih0h, G_DEC * DEC_HID);
  k_transpose_f16<<<blocks(DEC_HID * D_ATTN), 256, 0, stream>>>(Wk, WkTh, DEC_HID, D_ATTN);
  for (int dir = 0; dir < 2; ++dir) {
    k_transpose_f32<<<blocks(G_ENC * H_ENC), 256, 0, stream>>>(
        eWhh0 + (size_t)dir * G_ENC * H_ENC, WhhT0 + (size_t)dir * H_ENC * G_ENC, G_ENC, H_ENC);
    k_transpose_f32<<<blocks(G_ENC * H_ENC), 256, 0, stream>>>(
        eWhh1 + (size_t)dir * G_ENC * H_ENC, WhhT1 + (size_t)dir * H_ENC * G_ENC, G_ENC, H_ENC);
  }
  k_transpose_f32<<<blocks(G_DEC * DEC_HID), 256, 0, stream>>>(
      dWih + (size_t)G_DEC * DEC_HID, dWi1T, G_DEC, DEC_HID);
  for (int l = 0; l < 2; ++l) {
    k_transpose_f32<<<blocks(G_DEC * DEC_HID), 256, 0, stream>>>(
        dWhh + (size_t)l * G_DEC * DEC_HID, dWhhT + (size_t)l * DEC_HID * G_DEC,
        G_DEC, DEC_HID);
  }

  // 2. encoder layer 0: hoisted input GEMMs (WMMA), then recurrent scan
  for (int dir = 0; dir < 2; ++dir) {
    int tiles = (SEQ / 16) * (G_ENC / 16);                 // 4096 waves
    k_wmma_gemm<<<tiles * 32 / 256, 256, 0, stream>>>(
        Xh, Wih0h + (size_t)dir * G_ENC * E_DIM, eb0 + (size_t)dir * G_ENC,
        Gx0 + (size_t)dir * SEQ * G_ENC, SEQ, G_ENC, E_DIM);
  }
  k_enc_scan<<<2, 1024, 0, stream>>>(Gx0, WhhT0, Yh, hinit, cinit);

  // 3. encoder layer 1
  for (int dir = 0; dir < 2; ++dir) {
    int tiles = (SEQ / 16) * (G_ENC / 16);
    k_wmma_gemm<<<tiles * 32 / 256, 256, 0, stream>>>(
        Yh, Wih1h + (size_t)dir * G_ENC * 512, eb1 + (size_t)dir * G_ENC,
        Gx1 + (size_t)dir * SEQ * G_ENC, SEQ, G_ENC, 512);
  }
  k_enc_scan<<<2, 1024, 0, stream>>>(Gx1, WhhT1, Ench, hinit + DEC_HID, cinit + DEC_HID);

  // 4. attention keys K = enc @ Wk  (hoisted out of the decoder scan)
  {
    int tiles = (SEQ / 16) * (D_ATTN / 16);                // 1024 waves
    k_wmma_gemm<<<tiles * 32 / 256, 256, 0, stream>>>(
        Ench, WkTh, nullptr, Kmat, SEQ, D_ATTN, DEC_HID);
  }

  // 5. decoder: teacher-forced input projection as GEMM, then scan
  k_build_dec_in<<<blocks(T_STEPS * 512), 256, 0, stream>>>(
      inputs, gold, sos, lat_tab, in_tab, Xh, Inp0);
  {
    int tiles = (T_STEPS / 16) * (G_DEC / 16);             // 512 waves
    k_wmma_gemm<<<tiles * 32 / 256, 256, 0, stream>>>(
        Inp0, dWih0h, db, Gd0, T_STEPS, G_DEC, DEC_HID);
  }
  k_decoder<<<1, 1024, 0, stream>>>(
      Gd0, dWi1T, dWhhT, db + G_DEC,
      Kmat, Wq, av, hinit, cinit, alens, out);
}
